// Net_11312943858306
// MI455X (gfx1250) — matrix operand order repaired and checked by reading, not compiled, not run on hardware
//
#include <hip/hip_runtime.h>
#include <hip/hip_bf16.h>

#define N_CAM 4
#define N_PTS 8192

typedef __attribute__((ext_vector_type(16))) __bf16 v16bf;
typedef __attribute__((ext_vector_type(8)))  float  v8f;

#define WMMA_BF16(a, b, c) __builtin_amdgcn_wmma_f32_16x16x32_bf16( \
        false, (a), false, (b), (short)0, (c), false, false)

// ---------------------------------------------------------------------------
// branch-free guarded load: unconditional vmem + v_cndmask (no exec juggling)
// ---------------------------------------------------------------------------
__device__ inline float selload(const float* __restrict__ p, size_t idx, bool ok) {
    size_t ci = ok ? idx : 0;
    float v = p[ci];
    return ok ? v : 0.0f;
}

// ---------------------------------------------------------------------------
// WMMA fragment loaders (contiguous k order, used alike for A and B; wave32)
// A 16x32 bf16: lane<16: M=lane, K=k0..k0+15
//               lane>=16: M=lane-16, K=k0+16..k0+31
// B 32x16 bf16 (weights [N,K] row-major): lane<16: N=lane, K=k0..k0+15
//               lane>=16: K=k0+16..k0+31
// ---------------------------------------------------------------------------
__device__ inline v16bf load_a_fast4(const float* __restrict__ A, int K,
                                     int m0, int k0) {
    int lane = threadIdx.x & 31;
    int hi   = lane >> 4;
    const float* base = A + (size_t)(m0 + (lane & 15)) * K + k0 + (hi ? 16 : 0);
    float4 p0 = *(const float4*)(base);
    float4 p1 = *(const float4*)(base + 4);
    float4 p2 = *(const float4*)(base + 8);
    float4 p3 = *(const float4*)(base + 12);
    v16bf a;
    a[0]=(__bf16)p0.x; a[1]=(__bf16)p0.y; a[2]=(__bf16)p0.z; a[3]=(__bf16)p0.w;
    a[4]=(__bf16)p1.x; a[5]=(__bf16)p1.y; a[6]=(__bf16)p1.z; a[7]=(__bf16)p1.w;
    a[8]=(__bf16)p2.x; a[9]=(__bf16)p2.y; a[10]=(__bf16)p2.z; a[11]=(__bf16)p2.w;
    a[12]=(__bf16)p3.x; a[13]=(__bf16)p3.y; a[14]=(__bf16)p3.z; a[15]=(__bf16)p3.w;
    return a;
}

__device__ inline v16bf load_a_safe(const float* __restrict__ A, int M, int K,
                                    int m0, int k0) {
    int lane = threadIdx.x & 31;
    int hi   = lane >> 4;
    int m    = m0 + (lane & 15);
    v16bf a;
#pragma unroll
    for (int e = 0; e < 16; ++e) {
        int k = k0 + hi * 16 + e;
        bool ok = (m < M) && (k < K);
        a[e] = (__bf16)selload(A, (size_t)m * K + k, ok);
    }
    return a;
}

__device__ inline v16bf load_b_safe(const float* __restrict__ Wt, int N, int K,
                                    int n0, int k0) {
    int lane = threadIdx.x & 31;
    int hi   = lane >> 4;
    int n    = n0 + (lane & 15);
    v16bf b;
#pragma unroll
    for (int e = 0; e < 16; ++e) {
        int k = k0 + hi * 16 + e;
        bool ok = (n < N) && (k < K);
        b[e] = (__bf16)selload(Wt, (size_t)n * K + k, ok);
    }
    return b;
}

// implicit-GEMM A for 3x3 pad-1 conv on NCHW input (branch-free)
__device__ inline v16bf load_a_conv(const float* __restrict__ X, int Cin,
                                    int H, int W, int m0, int k0, int K) {
    int lane = threadIdx.x & 31;
    int hi   = lane >> 4;
    int m    = m0 + (lane & 15);
    int M    = N_CAM * H * W;
    int mc   = (m < M) ? m : 0;
    int w = mc % W; int t = mc / W; int h = t % H; int n = t / H;
    v16bf a;
#pragma unroll
    for (int e = 0; e < 16; ++e) {
        int k = k0 + hi * 16 + e;
        int kc = (k < K) ? k : 0;
        int ci = kc / 9, r = kc % 9;
        int kh = r / 3, kw = r % 3;
        int ih = h + kh - 1, iw = w + kw - 1;
        bool ok = (m < M) && (k < K) && (ih >= 0) && (ih < H) && (iw >= 0) && (iw < W);
        size_t idx = (((size_t)n * Cin + ci) * H + (ih >= 0 ? ih : 0)) * W + (iw >= 0 ? iw : 0);
        a[e] = (__bf16)selload(X, idx, ok);
    }
    return a;
}

// ---------------------------------------------------------------------------
// GEMM: C[M,N] = act(A[M,K] @ Wt[N,K]^T + bias[N])
// 4 waves / block, 64x32 tile. B tile staged in LDS as bf16 and shared by all
// waves (ds_load_b128 fragment reads); each wave owns a 16-row A stripe.
// ---------------------------------------------------------------------------
__global__ __launch_bounds__(128)
void gemm_wmma(const float* __restrict__ A, const float* __restrict__ Wt,
               const float* __restrict__ bias, float* __restrict__ C,
               int M, int N, int K, int relu) {
    __shared__ __align__(32) __bf16 Bs[32][32];     // [n][k], rows k-contiguous

    int tid  = threadIdx.x;
    int wave = tid >> 5;
    int lane = tid & 31;
    int hi   = lane >> 4;
    int nn   = lane & 15;

    int tmb = blockIdx.x * 64;          // block M origin
    int tm  = tmb + wave * 16;          // this wave's M origin
    int tn  = blockIdx.y * 32;          // block N origin

    // B-tile staging coords: thread -> 1 row x 8 k of the 32x32 tile
    int nr = tid >> 2;                  // 0..31
    int kq = (tid & 3) * 8;             // 0,8,16,24

    v8f acc0 = {}, acc1 = {};
    int  Kmain = K & ~31;
    bool fastT = (tmb + 64 <= M) && (tn + 32 <= N) && ((K & 3) == 0);

    if (fastT) {   // gfx1250 prefetch of this wave's A rows, once per tile
        __builtin_prefetch(&A[(size_t)(tm + nn) * K], 0, 0);
    }

    for (int k0 = 0; k0 < Kmain; k0 += 32) {
        if (fastT) {
            const float* src = Wt + (size_t)(tn + nr) * K + k0 + kq;
            float4 q0 = *(const float4*)(src);
            float4 q1 = *(const float4*)(src + 4);
            __bf16 tmp[8];
            tmp[0]=(__bf16)q0.x; tmp[1]=(__bf16)q0.y; tmp[2]=(__bf16)q0.z; tmp[3]=(__bf16)q0.w;
            tmp[4]=(__bf16)q1.x; tmp[5]=(__bf16)q1.y; tmp[6]=(__bf16)q1.z; tmp[7]=(__bf16)q1.w;
            __builtin_memcpy(&Bs[nr][kq], tmp, 16);
        } else {
            int n = tn + nr;
            __bf16 tmp[8];
#pragma unroll
            for (int j = 0; j < 8; ++j) {
                int k = k0 + kq + j;
                bool ok = (n < N) && (k < K);
                tmp[j] = (__bf16)selload(Wt, (size_t)n * K + k, ok);
            }
            __builtin_memcpy(&Bs[nr][kq], tmp, 16);
        }
        __syncthreads();

        v16bf a = fastT ? load_a_fast4(A, K, tm, k0)
                        : load_a_safe(A, M, K, tm, k0);
        v16bf b0 = *(const v16bf*)&Bs[nn][hi * 16];
        v16bf b1 = *(const v16bf*)&Bs[16 + nn][hi * 16];
        acc0 = WMMA_BF16(a, b0, acc0);
        acc1 = WMMA_BF16(a, b1, acc1);
        __syncthreads();
    }
    if (Kmain < K) {          // ragged K tail (branch-free guards)
        {
            int n = tn + nr;
            __bf16 tmp[8];
#pragma unroll
            for (int j = 0; j < 8; ++j) {
                int k = Kmain + kq + j;
                bool ok = (n < N) && (k < K);
                tmp[j] = (__bf16)selload(Wt, (size_t)n * K + k, ok);
            }
            __builtin_memcpy(&Bs[nr][kq], tmp, 16);
        }
        __syncthreads();
        v16bf a  = load_a_safe(A, M, K, tm, Kmain);
        v16bf b0 = *(const v16bf*)&Bs[nn][hi * 16];
        v16bf b1 = *(const v16bf*)&Bs[16 + nn][hi * 16];
        acc0 = WMMA_BF16(a, b0, acc0);
        acc1 = WMMA_BF16(a, b1, acc1);
        __syncthreads();
    }

    {
        int n = tn + nn;
        if (n < N) {
            float bv = bias[n];
#pragma unroll
            for (int r = 0; r < 8; ++r) {
                int m = tm + hi * 8 + r;
                if (m < M) {
                    float v = acc0[r] + bv;
                    if (relu) v = fmaxf(v, 0.0f);
                    C[(size_t)m * N + n] = v;
                }
            }
        }
    }
    {
        int n = tn + 16 + nn;
        if (n < N) {
            float bv = bias[n];
#pragma unroll
            for (int r = 0; r < 8; ++r) {
                int m = tm + hi * 8 + r;
                if (m < M) {
                    float v = acc1[r] + bv;
                    if (relu) v = fmaxf(v, 0.0f);
                    C[(size_t)m * N + n] = v;
                }
            }
        }
    }
}

// ---------------------------------------------------------------------------
// 3x3 pad-1 conv, implicit GEMM, output scattered to NCHW
// ---------------------------------------------------------------------------
__global__ __launch_bounds__(32)
void conv3x3_wmma(const float* __restrict__ X, const float* __restrict__ Wt,
                  const float* __restrict__ bias, float* __restrict__ Y,
                  int Cin, int Cout, int H, int W) {
    int M = N_CAM * H * W;
    int K = Cin * 9;
    int tm = blockIdx.x * 16;
    int tn = blockIdx.y * 16;
    v8f acc = {};
    for (int k0 = 0; k0 < K; k0 += 32) {
        v16bf a = load_a_conv(X, Cin, H, W, tm, k0, K);
        v16bf b = load_b_safe(Wt, Cout, K, tn, k0);
        acc = WMMA_BF16(a, b, acc);
    }
    int lane = threadIdx.x & 31;
    int hi   = lane >> 4;
    int co   = tn + (lane & 15);
    if (co >= Cout) return;
    float bv = bias[co];
#pragma unroll
    for (int r = 0; r < 8; ++r) {
        int m = tm + hi * 8 + r;
        if (m < M) {
            int w = m % W; int t = m / W; int h = t % H; int n = t / H;
            Y[(((size_t)n * Cout + co) * H + h) * W + w] = acc[r] + bv;
        }
    }
}

// ---------------------------------------------------------------------------
// BatchNorm: per-channel stats over (N,H,W) then in-place normalize (+ReLU)
// ---------------------------------------------------------------------------
__global__ void bn_stats(const float* __restrict__ X, float* __restrict__ mu,
                         float* __restrict__ istd, int C, int HW) {
    int c = blockIdx.x;
    size_t cnt = (size_t)N_CAM * HW;
    float s = 0.f, s2 = 0.f;
    for (size_t i = threadIdx.x; i < cnt; i += blockDim.x) {
        int n = (int)(i / HW), sp = (int)(i % HW);
        float v = X[((size_t)n * C + c) * HW + sp];
        s += v; s2 += v * v;
    }
    __shared__ float sh[256], sh2[256];
    sh[threadIdx.x] = s; sh2[threadIdx.x] = s2;
    __syncthreads();
    for (int st = blockDim.x / 2; st > 0; st >>= 1) {
        if ((int)threadIdx.x < st) {
            sh[threadIdx.x]  += sh[threadIdx.x + st];
            sh2[threadIdx.x] += sh2[threadIdx.x + st];
        }
        __syncthreads();
    }
    if (threadIdx.x == 0) {
        float n = (float)cnt;
        float m = sh[0] / n;
        float var = sh2[0] / n - m * m;
        mu[c] = m;
        istd[c] = rsqrtf(var + 1e-5f);
    }
}

__global__ void bn_apply(float* __restrict__ X, const float* __restrict__ mu,
                         const float* __restrict__ istd,
                         const float* __restrict__ gamma,
                         const float* __restrict__ beta,
                         int C, int HW, size_t total, int relu) {
    size_t i = blockIdx.x * (size_t)blockDim.x + threadIdx.x;
    if (i >= total) return;
    int c = (int)((i / HW) % C);
    float v = (X[i] - mu[c]) * istd[c] * gamma[c] + beta[c];
    if (relu) v = fmaxf(v, 0.0f);
    X[i] = v;
}

// ---------------------------------------------------------------------------
// 2x2 maxpool (stride 2), NCHW
// ---------------------------------------------------------------------------
__global__ void maxpool2(const float* __restrict__ X, float* __restrict__ Y,
                         int C, int H, int W, size_t total) {
    size_t i = blockIdx.x * (size_t)blockDim.x + threadIdx.x;
    if (i >= total) return;
    int Ho = H >> 1, Wo = W >> 1;
    int wo = (int)(i % Wo); size_t t = i / Wo;
    int ho = (int)(t % Ho); t /= Ho;
    int c  = (int)(t % C);  int n = (int)(t / C);
    const float* b = X + (((size_t)n * C + c) * H + 2 * ho) * W + 2 * wo;
    Y[i] = fmaxf(fmaxf(b[0], b[1]), fmaxf(b[W], b[W + 1]));
}

// ---------------------------------------------------------------------------
// Bilinear gather (replicates reference semantics incl. inside-masking and
// numpy-style negative-index wrap of the floor coordinate)
// ---------------------------------------------------------------------------
__global__ void gather_bilinear(const float* __restrict__ X,
                                const float* __restrict__ proj,
                                float* __restrict__ Y,
                                int C, int Hd, float inv, int off, int YC,
                                size_t total) {
    size_t i = blockIdx.x * (size_t)blockDim.x + threadIdx.x;
    if (i >= total) return;
    int c = (int)(i % C); size_t t = i / C;
    int pt = (int)(t % N_PTS); int cam = (int)(t / N_PTS);
    float px = proj[((size_t)cam * N_PTS + pt) * 2 + 0] * inv;
    float py = proj[((size_t)cam * N_PTS + pt) * 2 + 1] * inv;
    float xc = ceilf(px), xf = xc - 1.0f;
    float yc = ceilf(py), yf = yc - 1.0f;
    float lim = (float)(Hd - 1);
    float inside = (px <= lim && px >= 0.0f && py <= lim && py >= 0.0f) ? 1.0f : 0.0f;
    xc *= inside; xf *= inside; yc *= inside; yf *= inside;
    int xic = (int)xc, xif = (int)xf, yic = (int)yc, yif = (int)yf;
    if (xif < 0) xif += Hd;
    if (yif < 0) yif += Hd;
    const float* xb = X + ((size_t)cam * C + c) * Hd * Hd;
    float vcc = xb[(size_t)xic * Hd + yic];
    float vfc = xb[(size_t)xif * Hd + yic];
    float vcf = xb[(size_t)xic * Hd + yif];
    float vff = xb[(size_t)xif * Hd + yif];
    float wcc = (xf - px) * (yf - py);
    float wfc = (px - xc) * (yf - py);
    float wcf = (xf - px) * (py - yc);
    float wff = (px - xc) * (py - yc);
    Y[((size_t)cam * N_PTS + pt) * YC + off + c] =
        vcc * wcc + vfc * wfc + vcf * wcf + vff * wff;
}

// ---------------------------------------------------------------------------
// Cross-camera pool: z = concat(max_cam(h3), mean_cam(h3)), h3:[4,8192,1024]
// ---------------------------------------------------------------------------
__global__ void campool(const float* __restrict__ H3, float* __restrict__ Z,
                        size_t total) {
    size_t i = blockIdx.x * (size_t)blockDim.x + threadIdx.x;
    if (i >= total) return;                      // total = 8192*2048
    int c = (int)(i % 2048); int pt = (int)(i / 2048);
    if (c < 1024) {
        float v = H3[((size_t)0 * N_PTS + pt) * 1024 + c];
        for (int cam = 1; cam < N_CAM; ++cam)
            v = fmaxf(v, H3[((size_t)cam * N_PTS + pt) * 1024 + c]);
        Z[i] = v;
    } else {
        int cc = c - 1024;
        float s = 0.f;
        for (int cam = 0; cam < N_CAM; ++cam)
            s += H3[((size_t)cam * N_PTS + pt) * 1024 + cc];
        Z[i] = s * 0.25f;
    }
}

// ---------------------------------------------------------------------------
// utility: zero-fill, and zero-padded weight repack ([N,K] -> [N,Kp])
// ---------------------------------------------------------------------------
__global__ void fillzero(float* __restrict__ p, size_t n) {
    size_t i = blockIdx.x * (size_t)blockDim.x + threadIdx.x;
    if (i < n) p[i] = 0.0f;
}

__global__ void pad_weights(const float* __restrict__ W, float* __restrict__ Wp,
                            int K, int Kp, size_t total) {
    size_t i = blockIdx.x * (size_t)blockDim.x + threadIdx.x;
    if (i >= total) return;                      // total = N*Kp
    int k = (int)(i % Kp);
    size_t n = i / Kp;
    Wp[i] = (k < K) ? W[n * K + k] : 0.0f;
}

// ---------------------------------------------------------------------------
// Host orchestration
// ---------------------------------------------------------------------------
extern "C" void kernel_launch(void* const* d_in, const int* in_sizes, int n_in,
                              void* d_out, int out_size, void* d_ws, size_t ws_size,
                              hipStream_t stream) {
    const float* photo = (const float*)d_in[0];
    const float* proj  = (const float*)d_in[1];
    const float *cw[13], *cb[13], *bg[13], *bb[13], *fw[6], *fb[6];
    for (int i = 0; i < 13; ++i) { cw[i] = (const float*)d_in[2 + 2*i];  cb[i] = (const float*)d_in[3 + 2*i]; }
    for (int i = 0; i < 13; ++i) { bg[i] = (const float*)d_in[28 + 2*i]; bb[i] = (const float*)d_in[29 + 2*i]; }
    for (int i = 0; i < 6;  ++i) { fw[i] = (const float*)d_in[54 + 2*i]; fb[i] = (const float*)d_in[55 + 2*i]; }

    // workspace layout (fp32)
    float* wsf = (float*)d_ws;
    size_t off64 = 0;
    auto walloc = [&](size_t n) { float* p = wsf + off64; off64 += n; return p; };
    const size_t ACT = (size_t)N_CAM * 8 * 512 * 512;       // largest conv act
    const int    YK  = 1024;                                // padded feature K
    float* bufX  = walloc(ACT);
    float* bufT  = walloc(ACT);
    float* bufU  = walloc(ACT);
    float* mu    = walloc(512);
    float* istd  = walloc(512);
    float* y     = walloc((size_t)N_CAM * N_PTS * YK);      // padded features
    float* wpad0 = walloc((size_t)512 * YK);                // fc0 weights, K-padded
    float* h1    = walloc((size_t)N_CAM * N_PTS * 512);
    float* h3    = walloc((size_t)N_CAM * N_PTS * 1024);
    float* h2 = y;     // y dead after h1
    float* z  = h1;    // h1 dead after h2
    float* z1 = bufX;  // conv acts dead after gathers
    float* z2 = bufT;

    auto conv = [&](const float* X, float* Y, const float* w, const float* b,
                    int Cin, int Cout, int H) {
        int M = N_CAM * H * H;
        dim3 g((M + 15) / 16, (Cout + 15) / 16);
        conv3x3_wmma<<<g, 32, 0, stream>>>(X, w, b, Y, Cin, Cout, H, H);
    };
    auto bn = [&](float* X, const float* g_, const float* b_, int C, int H, int relu) {
        bn_stats<<<C, 256, 0, stream>>>(X, mu, istd, C, H * H);
        size_t total = (size_t)N_CAM * C * H * H;
        bn_apply<<<(int)((total + 255) / 256), 256, 0, stream>>>(
            X, mu, istd, g_, b_, C, H * H, total, relu);
    };
    auto gather = [&](const float* X, int C, int H, float inv, int featoff) {
        size_t total = (size_t)N_CAM * N_PTS * C;
        gather_bilinear<<<(int)((total + 255) / 256), 256, 0, stream>>>(
            X, proj, y, C, H, inv, featoff, YK, total);
    };
    auto gemm = [&](const float* A, const float* Wt, const float* bias, float* C,
                    int M, int N, int K, int relu) {
        dim3 g((M + 63) / 64, (N + 31) / 32);
        gemm_wmma<<<g, 128, 0, stream>>>(A, Wt, bias, C, M, N, K, relu);
    };

    // zero feature pad columns + build K-padded fc0 weights (runs every call)
    {
        size_t yt = (size_t)N_CAM * N_PTS * YK;
        fillzero<<<(int)((yt + 255) / 256), 256, 0, stream>>>(y, yt);
        size_t wt = (size_t)512 * YK;
        pad_weights<<<(int)((wt + 255) / 256), 256, 0, stream>>>(fw[0], wpad0, 1016, YK, wt);
    }

    // ---- backbone ----
    conv(photo, bufX, cw[0], cb[0], 3, 8, 512);
    bn(bufX, bg[0], bb[0], 8, 512, 0);
    gather(bufX, 8, 512, 1.0f, 0);

    const int stage_c[6][2] = {{8,16},{16,32},{32,64},{64,128},{128,256},{256,512}};
    int H = 512, Cc = 8, featoff = 8, ci = 1;
    float inv = 0.5f;
    for (int s = 0; s < 6; ++s) {
        int Ho = H >> 1;
        size_t pt_total = (size_t)N_CAM * Cc * Ho * Ho;
        maxpool2<<<(int)((pt_total + 255) / 256), 256, 0, stream>>>(
            bufX, bufT, Cc, H, H, pt_total);
        int C1 = stage_c[s][0], C2 = stage_c[s][1];
        conv(bufT, bufU, cw[ci], cb[ci], Cc, C1, Ho);
        bn(bufU, bg[ci], bb[ci], C1, Ho, 1);
        ci++;
        conv(bufU, bufX, cw[ci], cb[ci], C1, C2, Ho);
        bn(bufX, bg[ci], bb[ci], C2, Ho, 1);
        ci++;
        gather(bufX, C2, Ho, inv, featoff);
        featoff += C2; inv *= 0.5f; H = Ho; Cc = C2;
    }

    // ---- per-point MLP (WMMA GEMMs, dominant FLOPs; all K mult-of-32) ----
    const int MP = N_CAM * N_PTS;                        // 32768
    gemm(y,  wpad0, fb[0], h1, MP, 512,  YK,   1);
    gemm(h1, fw[1], fb[1], h2, MP, 512,  512,  1);
    gemm(h2, fw[2], fb[2], h3, MP, 1024, 512,  0);

    // ---- cross-camera pool + head ----
    size_t zt = (size_t)N_PTS * 2048;
    campool<<<(int)((zt + 255) / 256), 256, 0, stream>>>(h3, z, zt);
    gemm(z,  fw[3], fb[3], z1, N_PTS, 512, 2048, 1);
    gemm(z1, fw[4], fb[4], z2, N_PTS, 128, 512,  1);
    gemm(z2, fw[5], fb[5], (float*)d_out, N_PTS, 2, 128, 0);

    (void)in_sizes; (void)n_in; (void)out_size; (void)ws_size;
}
